// SOM_74955769250808
// MI455X (gfx1250) — matrix operand order repaired and checked by reading, not compiled, not run on hardware
//
#include <hip/hip_runtime.h>
#include <hip/hip_bf16.h>

// ---------------------------------------------------------------------------
// SOM update step for MI455X (gfx1250), wave32 + WMMA bf16.
//   M=N=64, GRID=4096, DIM=256, BATCH=4096
// Outputs (concat, float): new_centroids[4096*256], bmu_loc[4096*2], mean_mindist[1]
//
// Strategy (roofline): both GEMMs ([4096x4096x256] each, ~17 GFLOP total) run
// on v_wmma_f32_16x16x32_bf16. distance_mat (64 MB) is never read: the
// neighbourhood Gaussian is recomputed analytically from BMU grid coords.
// A prep pass writes operand-ready bf16 copies of x / centroids (4 MB, L2-
// resident at 192 MB), so the WMMA hot loops issue pure b128 bf16 loads with
// zero conversions per WMMA.
// ---------------------------------------------------------------------------

#define DIMK   256
#define GRIDN  4096
#define BATCHN 4096
#define ALPHA_C 0.5f
#define SIGMA_C 32.0f

typedef __attribute__((ext_vector_type(16))) __bf16 v16bf;
typedef __attribute__((ext_vector_type(8)))  __bf16 v8bf;
typedef __attribute__((ext_vector_type(8)))  float  v8f;

// ds_swizzle XOR butterfly (group-of-32 mode: offset = (xor<<10) | 0x1F)
template<int OFF>
__device__ __forceinline__ float swzf(float v) {
    return __int_as_float(__builtin_amdgcn_ds_swizzle(__float_as_int(v), OFF));
}
template<int OFF>
__device__ __forceinline__ int swzi(int v) {
    return __builtin_amdgcn_ds_swizzle(v, OFF);
}
#define SWZ_X1  0x041F
#define SWZ_X2  0x081F
#define SWZ_X4  0x101F
#define SWZ_X8  0x201F
#define SWZ_X16 0x401F

__device__ __forceinline__ v16bf cat16(v8bf lo, v8bf hi) {
    v16bf r;
#pragma unroll
    for (int e = 0; e < 8; ++e) { r[e] = lo[e]; r[8 + e] = hi[e]; }
    return r;
}

// ---------------------------------------------------------------------------
// Kernel 0 (prep): one wave per row (x rows then centroid rows).
//  - squared norm of the row  -> xnorm / cnorm
//  - bf16 copy of the row     -> xbf / cbf  (row-major, b128 stores)
// ---------------------------------------------------------------------------
__global__ __launch_bounds__(256) void k_prep(const float* __restrict__ x,
                                              const float* __restrict__ cent,
                                              float* __restrict__ xnorm,
                                              float* __restrict__ cnorm,
                                              __bf16* __restrict__ xbf,
                                              __bf16* __restrict__ cbf) {
    int wid  = (blockIdx.x * blockDim.x + threadIdx.x) >> 5;   // row id 0..8191
    int lane = threadIdx.x & 31;
    const bool  isX = (wid < BATCHN);
    const float* p  = isX ? (x + (size_t)wid * DIMK)
                          : (cent + (size_t)(wid - BATCHN) * DIMK);
    __bf16* q = isX ? (xbf + (size_t)wid * DIMK)
                    : (cbf + (size_t)(wid - BATCHN) * DIMK);

    float s = 0.f;
    const int col = lane * 8;
    float4 a0 = *(const float4*)(p + col);
    float4 a1 = *(const float4*)(p + col + 4);
    s += a0.x * a0.x + a0.y * a0.y + a0.z * a0.z + a0.w * a0.w;
    s += a1.x * a1.x + a1.y * a1.y + a1.z * a1.z + a1.w * a1.w;
    v8bf b;
    b[0] = (__bf16)a0.x; b[1] = (__bf16)a0.y; b[2] = (__bf16)a0.z; b[3] = (__bf16)a0.w;
    b[4] = (__bf16)a1.x; b[5] = (__bf16)a1.y; b[6] = (__bf16)a1.z; b[7] = (__bf16)a1.w;
    *(v8bf*)(q + col) = b;

    s += swzf<SWZ_X1>(s);
    s += swzf<SWZ_X2>(s);
    s += swzf<SWZ_X4>(s);
    s += swzf<SWZ_X8>(s);
    s += swzf<SWZ_X16>(s);
    if (lane == 0) {
        if (isX) xnorm[wid] = s;
        else     cnorm[wid - BATCHN] = s;
    }
}

// ---------------------------------------------------------------------------
// Kernel 1: BMU search.  S = x @ cent^T via v_wmma_f32_16x16x32_bf16,
// running argmin over (cnorm[g] - 2*S) (x-norm added only at the end).
// Block: 256 thr = 8 waves, each wave owns 16 batch rows -> 128 rows/block.
// Operands come pre-converted from xbf/cbf: the hot loop is b128 loads + WMMA.
// ---------------------------------------------------------------------------
__global__ __launch_bounds__(256) void k_bmu(const __bf16* __restrict__ xbf,
                                             const __bf16* __restrict__ cbf,
                                             const float* __restrict__ xnorm,
                                             const float* __restrict__ cnorm,
                                             float* __restrict__ ws_mind,
                                             float* __restrict__ ws_bi,
                                             float* __restrict__ ws_bj,
                                             float* __restrict__ out_bmu) {
    const int wave = threadIdx.x >> 5;
    const int lane = threadIdx.x & 31;
    const int half = lane >> 4;          // 0: lanes 0-15, 1: lanes 16-31
    const int l15  = lane & 15;
    const int rowBase = blockIdx.x * 128 + wave * 16;

    // ---- preload A (16 rows x 256 K) as 8 bf16 chunks in WMMA A-layout ----
    // lane l: M = l&15 ; elems 0..7 -> K = 32c + 8*half + e ; 8..15 -> +16
    v16bf A[8];
    const __bf16* xp = xbf + (size_t)(rowBase + l15) * DIMK;
#pragma unroll
    for (int c = 0; c < 8; ++c) {
        const int s1 = c * 32 + half * 8;
        v8bf lo = *(const v8bf*)(xp + s1);
        v8bf hi = *(const v8bf*)(xp + s1 + 16);
        A[c] = cat16(lo, hi);
    }

    float minv[8];
    int   mincol[8];
#pragma unroll
    for (int v = 0; v < 8; ++v) { minv[v] = 3.0e38f; mincol[v] = 0; }

    // ---- sweep all 256 centroid column tiles ----
    for (int ct = 0; ct < GRIDN / 16; ++ct) {
        const int colBase = ct * 16;
        // B-layout: lane l holds column n = l&15, elems 0..7 -> K = 32c + 8*half + e ; 8..15 -> +16
        const __bf16* cp = cbf + (size_t)(colBase + l15) * DIMK + half * 8;
        if (ct + 1 < GRIDN / 16) {
            __builtin_prefetch(cp + 16 * DIMK, 0, 1);   // next tile, same lane row
        }
        v8f acc = {0.f, 0.f, 0.f, 0.f, 0.f, 0.f, 0.f, 0.f};
#pragma unroll
        for (int c = 0; c < 8; ++c) {
            v8bf lo = *(const v8bf*)(cp + c * 32);
            v8bf hi = *(const v8bf*)(cp + c * 32 + 16);
            acc = __builtin_amdgcn_wmma_f32_16x16x32_bf16(
                false, A[c], false, cat16(lo, hi), (short)0, acc, false, false);
        }
        const float cn  = cnorm[colBase + l15];
        const int   col = colBase + l15;
#pragma unroll
        for (int v = 0; v < 8; ++v) {
            float s = cn - 2.0f * acc[v];
            if (s < minv[v]) { minv[v] = s; mincol[v] = col; }
        }
    }

    // ---- cross-lane argmin over the 16 lanes that share each row ----
#pragma unroll
    for (int v = 0; v < 8; ++v) {
        {
            float ov = swzf<SWZ_X1>(minv[v]); int oc = swzi<SWZ_X1>(mincol[v]);
            if (ov < minv[v] || (ov == minv[v] && oc < mincol[v])) { minv[v] = ov; mincol[v] = oc; }
        }
        {
            float ov = swzf<SWZ_X2>(minv[v]); int oc = swzi<SWZ_X2>(mincol[v]);
            if (ov < minv[v] || (ov == minv[v] && oc < mincol[v])) { minv[v] = ov; mincol[v] = oc; }
        }
        {
            float ov = swzf<SWZ_X4>(minv[v]); int oc = swzi<SWZ_X4>(mincol[v]);
            if (ov < minv[v] || (ov == minv[v] && oc < mincol[v])) { minv[v] = ov; mincol[v] = oc; }
        }
        {
            float ov = swzf<SWZ_X8>(minv[v]); int oc = swzi<SWZ_X8>(mincol[v]);
            if (ov < minv[v] || (ov == minv[v] && oc < mincol[v])) { minv[v] = ov; mincol[v] = oc; }
        }
    }

    if (l15 == 0) {
#pragma unroll
        for (int v = 0; v < 8; ++v) {
            const int row = rowBase + v + half * 8;   // D-layout M mapping
            float d2   = xnorm[row] + minv[v];
            float dist = sqrtf(fmaxf(d2, 0.0f));
            int bmu = mincol[v];
            float bi = (float)(bmu >> 6);
            float bj = (float)(bmu & 63);
            ws_mind[row] = dist;
            ws_bi[row]   = bi;
            ws_bj[row]   = bj;
            out_bmu[row * 2 + 0] = bi;
            out_bmu[row * 2 + 1] = bj;
        }
    }
}

// ---------------------------------------------------------------------------
// Kernel 2: mean(mindist) -> single float.
// ---------------------------------------------------------------------------
__global__ __launch_bounds__(256) void k_mindist(const float* __restrict__ mind,
                                                 float* __restrict__ out) {
    __shared__ float red[8];
    float s = 0.f;
    for (int i = threadIdx.x; i < BATCHN; i += 256) s += mind[i];
    s += swzf<SWZ_X1>(s);
    s += swzf<SWZ_X2>(s);
    s += swzf<SWZ_X4>(s);
    s += swzf<SWZ_X8>(s);
    s += swzf<SWZ_X16>(s);
    int lane = threadIdx.x & 31, wave = threadIdx.x >> 5;
    if (lane == 0) red[wave] = s;
    __syncthreads();
    if (threadIdx.x == 0) {
        float t = 0.f;
#pragma unroll
        for (int i = 0; i < 8; ++i) t += red[i];
        *out = t * (1.0f / (float)BATCHN);
    }
}

// ---------------------------------------------------------------------------
// Kernel 3: centroid update.
//   acc[g,d] = sum_b lrm[b,g] * x[b,d]   (WMMA bf16, K = BATCH = 4096)
//   lrm computed analytically from BMU grid coords (no distance_mat traffic)
//   new_c = c*(1 - rowsum/B) + acc/B
// Block: 256 thr = 8 waves; wave w -> M-tile (w>>1), N-half (w&1).
// x chunk (32 rows x 256 dims, already bf16) staged TRANSPOSED in LDS so the
// B-operand per lane is two contiguous ds_load_b128 of ready bf16.
// ---------------------------------------------------------------------------
__global__ __launch_bounds__(256) void k_update(const __bf16* __restrict__ xbf,
                                                const float* __restrict__ cent,
                                                const float* __restrict__ ws_bi,
                                                const float* __restrict__ ws_bj,
                                                const float* __restrict__ lr_p,
                                                float* __restrict__ out) {
    __shared__ __bf16 xs[32 * DIMK];   // 16 KB: x chunk, stored [d][b]

    const int wave = threadIdx.x >> 5;
    const int lane = threadIdx.x & 31;
    const int half = lane >> 4;
    const int l15  = lane & 15;
    const int mt   = wave >> 1;       // M-tile within block (0..3)
    const int nh   = wave & 1;        // N half: dims [nh*128, nh*128+128)
    const int gBase = blockIdx.x * 64 + mt * 16;

    const float lr       = lr_p[0];
    const float alpha_op = ALPHA_C * lr;
    const float sigma_op = SIGMA_C * lr;
    const float invDen   = 1.0f / (2.0f * sigma_op * sigma_op + 1e-5f);
    const float invB     = 1.0f / (float)BATCHN;

    const int g  = gBase + l15;       // A-layout row for this lane
    const float gi = (float)(g >> 6);
    const float gj = (float)(g & 63);

    v8f acc[8];
#pragma unroll
    for (int t = 0; t < 8; ++t) acc[t] = (v8f){0.f,0.f,0.f,0.f,0.f,0.f,0.f,0.f};
    float rowsum = 0.f;

    for (int kc = 0; kc < BATCHN / 32; ++kc) {
        const int b0 = kc * 32;

        // ---- cooperative transpose-stage of xbf[b0..b0+31][0..255] ----
        const v8bf* xsrc = (const v8bf*)(xbf + (size_t)b0 * DIMK);
#pragma unroll
        for (int i = 0; i < 4; ++i) {
            int j = threadIdx.x + i * 256;     // v8bf index within chunk (0..1023)
            v8bf v = xsrc[j];
            int brow = j >> 5;                 // 32 v8bf per source row
            int d0   = (j & 31) * 8;
#pragma unroll
            for (int e = 0; e < 8; ++e) xs[(d0 + e) * 32 + brow] = v[e];
        }
        __syncthreads();

        // ---- A tile on the fly: lrm[b,g] = alpha * exp(-dist2 * invDen) ----
        const int r0 = b0 + half * 8;          // elems 0..7 -> b = r0+e ; 8..15 -> r0+16+e
        float4 i0 = *(const float4*)(ws_bi + r0);
        float4 i1 = *(const float4*)(ws_bi + r0 + 4);
        float4 i2 = *(const float4*)(ws_bi + r0 + 16);
        float4 i3 = *(const float4*)(ws_bi + r0 + 20);
        float4 j0 = *(const float4*)(ws_bj + r0);
        float4 j1 = *(const float4*)(ws_bj + r0 + 4);
        float4 j2 = *(const float4*)(ws_bj + r0 + 16);
        float4 j3 = *(const float4*)(ws_bj + r0 + 20);
        float bi[16] = {i0.x,i0.y,i0.z,i0.w, i1.x,i1.y,i1.z,i1.w,
                        i2.x,i2.y,i2.z,i2.w, i3.x,i3.y,i3.z,i3.w};
        float bj[16] = {j0.x,j0.y,j0.z,j0.w, j1.x,j1.y,j1.z,j1.w,
                        j2.x,j2.y,j2.z,j2.w, j3.x,j3.y,j3.z,j3.w};
        v16bf Abf;
#pragma unroll
        for (int e = 0; e < 16; ++e) {
            float di = gi - bi[e];
            float dj = gj - bj[e];
            float lv = alpha_op * __expf(-(di * di + dj * dj) * invDen);
            rowsum += lv;
            Abf[e] = (__bf16)lv;
        }

        // ---- 8 N-tiles: B from LDS (two contiguous ds_load_b128 per lane) ----
#pragma unroll
        for (int nt8 = 0; nt8 < 8; ++nt8) {
            const int d = nh * 128 + nt8 * 16 + l15;
            const v8bf* bp = (const v8bf*)&xs[d * 32 + half * 8];
            v8bf lo = bp[0];
            v8bf hi = bp[2];
            acc[nt8] = __builtin_amdgcn_wmma_f32_16x16x32_bf16(
                false, Abf, false, cat16(lo, hi), (short)0, acc[nt8], false, false);
        }
        __syncthreads();
    }

    // ---- rowsum: pair-combine halves, then bpermute to D-layout rows ----
    float rsF = rowsum + swzf<SWZ_X16>(rowsum);   // full sum for g = gBase + l15

#pragma unroll
    for (int v = 0; v < 8; ++v) {
        const int r  = v + half * 8;              // D-layout row index
        const float rs = __int_as_float(
            __builtin_amdgcn_ds_bpermute(r << 2, __float_as_int(rsF)));
        const int grow = gBase + r;
        const float scale = 1.0f - rs * invB;
#pragma unroll
        for (int nt8 = 0; nt8 < 8; ++nt8) {
            const int d = nh * 128 + nt8 * 16 + l15;
            const size_t idx = (size_t)grow * DIMK + d;
            float c = cent[idx];
            out[idx] = c * scale + acc[nt8][v] * invB;
        }
    }
}

// ---------------------------------------------------------------------------
// Host-side launcher
// ---------------------------------------------------------------------------
extern "C" void kernel_launch(void* const* d_in, const int* in_sizes, int n_in,
                              void* d_out, int out_size, void* d_ws, size_t ws_size,
                              hipStream_t stream) {
    const float* x    = (const float*)d_in[0];   // [BATCH, DIM]
    const float* cent = (const float*)d_in[1];   // [GRID, DIM]
    // d_in[2] = distance_mat (recomputed analytically), d_in[3] = locations (implied)
    const float* lr   = (const float*)d_in[4];   // scalar

    float* out = (float*)d_out;
    float* out_centroids = out;                          // GRID*DIM
    float* out_bmu       = out + (size_t)GRIDN * DIMK;   // BATCH*2
    float* out_mind      = out_bmu + (size_t)BATCHN * 2; // 1

    // ws layout: 5*4096 f32 (80 KB, 256B-aligned end) + bf16 x (2 MB) + bf16 cent (2 MB)
    float* ws      = (float*)d_ws;
    float* ws_xn   = ws;
    float* ws_cn   = ws + BATCHN;
    float* ws_mind = ws + BATCHN + GRIDN;
    float* ws_bi   = ws + BATCHN + GRIDN + BATCHN;
    float* ws_bj   = ws + BATCHN + GRIDN + 2 * BATCHN;
    __bf16* xbf    = (__bf16*)((char*)d_ws + 81920);
    __bf16* cbf    = xbf + (size_t)BATCHN * DIMK;

    k_prep   <<<(BATCHN + GRIDN) * 32 / 256, 256, 0, stream>>>(x, cent, ws_xn, ws_cn, xbf, cbf);
    k_bmu    <<<BATCHN / 128, 256, 0, stream>>>(xbf, cbf, ws_xn, ws_cn,
                                                ws_mind, ws_bi, ws_bj, out_bmu);
    k_mindist<<<1, 256, 0, stream>>>(ws_mind, out_mind);
    k_update <<<GRIDN / 64, 256, 0, stream>>>(xbf, cent, ws_bi, ws_bj, lr, out_centroids);
    (void)in_sizes; (void)n_in; (void)out_size; (void)ws_size;
}
